// PCTClassifier_43413529428106
// MI455X (gfx1250) — matrix operand order repaired and checked by reading, not compiled, not run on hardware
//
#include <hip/hip_runtime.h>
#include <hip/hip_bf16.h>
#include <math.h>

// ---------------- types / constants ----------------
typedef __attribute__((ext_vector_type(16))) _Float16 v16h;
typedef __attribute__((ext_vector_type(8)))  _Float16 v8h;
typedef __attribute__((ext_vector_type(8)))  float    v8f;

#define BB   8
#define NPTS 4096
#define S1   2048
#define S2   1024
#define KNB  32
#define NCLS 40
#define RSQ  0.9999950000374997f   /* 1/sqrt(1+1e-5) */
#define CDIV(a,b) (((a)+(b)-1)/(b))

// input index map: d_in[0]=xyz, then params leaves with nested dict keys sorted
enum {
  I_XYZ = 0,
  I_B1_B, I_B1_G, I_B1_W,
  I_B2_B, I_B2_G, I_B2_W,
  I_FC1_B, I_FC1_BIAS, I_FC1_G, I_FC1_W,
  I_FC2_B, I_FC2_BIAS, I_FC2_G, I_FC2_W,
  I_FC3_BIAS, I_FC3_W,
  I_FUSE_B, I_FUSE_G, I_FUSE_W,
  I_OA1 = 20,   // +0 proj.b +1 proj.g +2 proj.w +3 wk +4 wq +5 wv
  I_SG1_L1_B = 44, I_SG1_L1_G, I_SG1_L1_W,
  I_SG1_L2_B, I_SG1_L2_G, I_SG1_L2_W,
  I_SG2_L1_B = 50, I_SG2_L1_G, I_SG2_L1_W,
  I_SG2_L2_B, I_SG2_L2_G, I_SG2_L2_W
};

// ---------------- WMMA GEMM: Y = act((X @ W^T)*scale + bias) (+res) ----------------
// X f16 [M,Kd], W f16 [O,Kd] (row major), Y f16 [M,O]. Optional batched via blockIdx.z.
// Each wave computes a 32x32 output tile (2x2 grid of 16x16 WMMA tiles): per K-step
// 2 A-fragments + 2 B-fragments feed 4 independent v_wmma_f32_16x16x32_f16, halving
// loads-per-wmma vs a 16x16 tile and avoiding WMMA->WMMA RAW hazards. 8 waves/block
// tile M (256 rows/block). Epilogue folds BN (gamma*RSQ, beta), optional ReLU, optional
// f16 residual (ld = O). All M and O used here are multiples of 32.
__global__ void __launch_bounds__(256)
k_wmma_gemm(const _Float16* __restrict__ X, const _Float16* __restrict__ W,
            _Float16* __restrict__ Y,
            const float* __restrict__ gamma, const float* __restrict__ beta,
            const _Float16* __restrict__ res,
            int M, int Kd, int O,
            long long sX, long long sW, long long sY, long long sRes, int relu)
{
  const int lane = threadIdx.x & 31;
  const int wave = threadIdx.x >> 5;
  const int hl   = lane >> 4;      // half of wave (0: lanes 0-15, 1: lanes 16-31)
  const int l16  = lane & 15;
  const int m0 = (blockIdx.x * 8 + wave) * 32;
  const int o0 = blockIdx.y * 32;
  if (m0 >= M) return;
  const long long zb = blockIdx.z;
  X += zb * sX; W += zb * sW; Y += zb * sY;

  // A fragment (16x32 f16): lane holds row; VGPR chunks: K = hl*8..+7 and 16+hl*8..+7
  const _Float16* xp0 = X + (size_t)(m0 + l16) * Kd + hl * 8;
  const _Float16* xp1 = xp0 + (size_t)16 * Kd;
  // B fragment (32x16 f16): lane holds column o; K = hl*8..+7 and 16+hl*8..+7
  const _Float16* wp0 = W + (size_t)(o0 + l16) * Kd + hl * 8;
  const _Float16* wp1 = wp0 + (size_t)16 * Kd;

  v8f acc[2][2] = {{{}, {}}, {{}, {}}};
  for (int k0 = 0; k0 < Kd; k0 += 32) {
    v8h a0lo = *(const v8h*)(xp0 + k0);
    v8h a0hi = *(const v8h*)(xp0 + k0 + 16);
    v8h a1lo = *(const v8h*)(xp1 + k0);
    v8h a1hi = *(const v8h*)(xp1 + k0 + 16);
    v8h b0lo = *(const v8h*)(wp0 + k0);
    v8h b0hi = *(const v8h*)(wp0 + k0 + 16);
    v8h b1lo = *(const v8h*)(wp1 + k0);
    v8h b1hi = *(const v8h*)(wp1 + k0 + 16);
    v16h a0, a1, b0, b1;
#pragma unroll
    for (int e = 0; e < 8; ++e) {
      a0[e] = a0lo[e]; a0[e + 8] = a0hi[e];
      a1[e] = a1lo[e]; a1[e + 8] = a1hi[e];
      b0[e] = b0lo[e]; b0[e + 8] = b0hi[e];
      b1[e] = b1lo[e]; b1[e + 8] = b1hi[e];
    }
    acc[0][0] = __builtin_amdgcn_wmma_f32_16x16x32_f16(false, a0, false, b0,
                                                       (short)0, acc[0][0], false, false);
    acc[0][1] = __builtin_amdgcn_wmma_f32_16x16x32_f16(false, a0, false, b1,
                                                       (short)0, acc[0][1], false, false);
    acc[1][0] = __builtin_amdgcn_wmma_f32_16x16x32_f16(false, a1, false, b0,
                                                       (short)0, acc[1][0], false, false);
    acc[1][1] = __builtin_amdgcn_wmma_f32_16x16x32_f16(false, a1, false, b1,
                                                       (short)0, acc[1][1], false, false);
  }

  const _Float16* rp = res ? (res + zb * sRes) : (const _Float16*)0;
#pragma unroll
  for (int ot = 0; ot < 2; ++ot) {
    const int o = o0 + 16 * ot + l16;
    const float sc = gamma ? gamma[o] * RSQ : 1.0f;
    const float bi = beta  ? beta[o] : 0.0f;
#pragma unroll
    for (int mt = 0; mt < 2; ++mt) {
#pragma unroll
      for (int i = 0; i < 8; ++i) {         // D layout: VGPR i -> M = i + 8*hl, N = l16
        int m = m0 + 16 * mt + hl * 8 + i;
        float v = acc[mt][ot][i] * sc + bi;
        if (relu) v = v > 0.f ? v : 0.f;
        if (rp) v += (float)rp[(size_t)m * O + o];
        if (m < M) Y[(size_t)m * O + o] = (_Float16)v;
      }
    }
  }
}

// ---------------- small per-point linear + BN + ReLU (f32 weights) ----------------
__global__ void k_lin_small(const _Float16* __restrict__ X, const float* __restrict__ W,
                            const float* __restrict__ g, const float* __restrict__ b,
                            _Float16* __restrict__ Y, int rows, int C, int O)
{
  int gid = blockIdx.x * blockDim.x + threadIdx.x;
  if (gid >= rows * O) return;
  int o = gid % O, r = gid / O;
  const _Float16* x = X + (size_t)r * C;
  const float* w = W + (size_t)o * C;
  float s = 0.f;
  for (int c = 0; c < C; ++c) s += (float)x[c] * w[c];
  s = s * (g[o] * RSQ) + b[o];
  Y[gid] = (_Float16)(s > 0.f ? s : 0.f);
}

// ---------------- f32 -> f16 ----------------
__global__ void k_cvt(const float* __restrict__ s, _Float16* __restrict__ d, int n)
{
  int i = blockIdx.x * blockDim.x + threadIdx.x;
  if (i < n) d[i] = (_Float16)s[i];
}

// ---------------- farthest point sampling: one block per batch ----------------
__global__ void k_fps(const float* __restrict__ xyz, int Np, int npoint,
                      int* __restrict__ out_idx, float* __restrict__ dist)
{
  int b = blockIdx.x, t = threadIdx.x, nt = blockDim.x;
  const float* X = xyz + (size_t)b * Np * 3;
  int*   idx = out_idx + (size_t)b * npoint;
  float* D   = dist    + (size_t)b * Np;
  __shared__ float s_val[256];
  __shared__ int   s_idx[256];
  for (int i = t; i < Np; i += nt) D[i] = 1e10f;
  __syncthreads();
  int far = 0;
  for (int it = 0; it < npoint; ++it) {
    if (t == 0) idx[it] = far;
    float cx = X[far*3], cy = X[far*3+1], cz = X[far*3+2];
    float best = -1.f; int bi = 0;
    for (int i = t; i < Np; i += nt) {
      float dx = X[i*3]-cx, dy = X[i*3+1]-cy, dz = X[i*3+2]-cz;
      float d = dx*dx + dy*dy + dz*dz;
      float od = D[i]; d = d < od ? d : od; D[i] = d;
      if (d > best) { best = d; bi = i; }
    }
    s_val[t] = best; s_idx[t] = bi;
    __syncthreads();
    for (int s = nt >> 1; s > 0; s >>= 1) {
      if (t < s) {
        if (s_val[t+s] > s_val[t] ||
            (s_val[t+s] == s_val[t] && s_idx[t+s] < s_idx[t])) {
          s_val[t] = s_val[t+s]; s_idx[t] = s_idx[t+s];
        }
      }
      __syncthreads();
    }
    far = s_idx[0];
    __syncthreads();
  }
}

// ---------------- KNN (k=32 smallest sq-dist), one thread per (b,s) ----------------
__global__ void k_knn(const float* __restrict__ xyz, const int* __restrict__ cent,
                      int* __restrict__ knn, int Np, int S, int Bt)
{
  int gid = blockIdx.x * blockDim.x + threadIdx.x;
  if (gid >= Bt * S) return;
  int b = gid / S;
  const float* X = xyz + (size_t)b * Np * 3;
  int ci = cent[gid];
  float cx = X[ci*3], cy = X[ci*3+1], cz = X[ci*3+2];
  float bd[KNB]; int bi[KNB];
  for (int j = 0; j < KNB; ++j) { bd[j] = 3.0e38f; bi[j] = 0; }
  float worst = 3.0e38f; int wj = 0;
  for (int i = 0; i < Np; ++i) {
    float dx = X[i*3]-cx, dy = X[i*3+1]-cy, dz = X[i*3+2]-cz;
    float d = dx*dx + dy*dy + dz*dz;
    if (d < worst) {
      bd[wj] = d; bi[wj] = i;
      worst = -1.f;
      for (int j = 0; j < KNB; ++j) if (bd[j] > worst) { worst = bd[j]; wj = j; }
    }
  }
  int* out = knn + (size_t)gid * KNB;
  for (int j = 0; j < KNB; ++j) out[j] = bi[j];
}

// ---------------- build edge features: [neigh-center, center], f16 ----------------
__global__ void k_edge(const _Float16* __restrict__ feat, const int* __restrict__ cent,
                       const int* __restrict__ knn, _Float16* __restrict__ edge,
                       int Np, int S, int C, int Bt)
{
  int gid = blockIdx.x * blockDim.x + threadIdx.x;
  if (gid >= Bt * S * KNB) return;
  int j = gid % KNB, s = (gid / KNB) % S, b = gid / (KNB * S);
  (void)j;
  const _Float16* fc = feat + ((size_t)b * Np + cent[b * S + s]) * C;
  const _Float16* fn = feat + ((size_t)b * Np + knn[(size_t)gid]) * C;
  _Float16* e = edge + (size_t)gid * 2 * C;
  for (int c = 0; c < C; ++c) {
    float cv = (float)fc[c], nv = (float)fn[c];
    e[c]     = (_Float16)(nv - cv);
    e[C + c] = (_Float16)cv;
  }
}

// ---------------- max over KNB neighbors ----------------
__global__ void k_maxk(const _Float16* __restrict__ src, _Float16* __restrict__ dst,
                       int R, int C)
{
  int gid = blockIdx.x * blockDim.x + threadIdx.x;
  if (gid >= R * C) return;
  int c = gid % C, r = gid / C;
  const _Float16* s = src + ((size_t)r * KNB) * C + c;
  float m = (float)s[0];
  for (int j = 1; j < KNB; ++j) { float v = (float)s[(size_t)j * C]; m = v > m ? v : m; }
  dst[gid] = (_Float16)m;
}

// ---------------- gather sampled xyz ----------------
__global__ void k_gather3(const float* __restrict__ xyz, const int* __restrict__ idx,
                          float* __restrict__ out, int Np, int S, int Bt)
{
  int gid = blockIdx.x * blockDim.x + threadIdx.x;
  if (gid >= Bt * S) return;
  int b = gid / S;
  const float* p = xyz + ((size_t)b * Np + idx[gid]) * 3;
  out[(size_t)gid*3] = p[0]; out[(size_t)gid*3+1] = p[1]; out[(size_t)gid*3+2] = p[2];
}

// ---------------- transpose per-batch: [b,R,C] -> [b,C,R] ----------------
__global__ void k_transpose(const _Float16* __restrict__ src, _Float16* __restrict__ dst,
                            int R, int C, int Bt)
{
  size_t gid = (size_t)blockIdx.x * blockDim.x + threadIdx.x;
  if (gid >= (size_t)Bt * R * C) return;
  int c = (int)(gid % C); size_t t = gid / C; int r = (int)(t % R); int b = (int)(t / R);
  dst[((size_t)b * C + c) * R + r] = src[gid];
}

// ---------------- softmax over query axis (column-wise) ----------------
__global__ void k_col_softmax(const _Float16* __restrict__ E, float* __restrict__ A,
                              int N, int Bt)
{
  int gid = blockIdx.x * blockDim.x + threadIdx.x;
  if (gid >= Bt * N) return;
  int m = gid % N, b = gid / N;
  const _Float16* e = E + (size_t)b * N * N + m;
  float mx = -3.0e38f;
  for (int n = 0; n < N; ++n) { float v = (float)e[(size_t)n * N]; mx = v > mx ? v : mx; }
  float s = 0.f;
  for (int n = 0; n < N; ++n) s += expf((float)e[(size_t)n * N] - mx);
  float* a = A + (size_t)b * N * N + m;
  float inv = 1.0f / s;
  for (int n = 0; n < N; ++n) a[(size_t)n * N] = expf((float)e[(size_t)n * N] - mx) * inv;
}

// ---------------- L1 renorm over key axis (row-wise), write f16 ----------------
__global__ void k_row_renorm(const float* __restrict__ A, _Float16* __restrict__ Ah,
                             int N, int Bt)
{
  int gid = blockIdx.x * blockDim.x + threadIdx.x;
  if (gid >= Bt * N) return;
  const float* r = A + (size_t)gid * N;
  float s = 0.f;
  for (int m = 0; m < N; ++m) s += r[m];
  float inv = 1.0f / (1e-9f + s);
  _Float16* o = Ah + (size_t)gid * N;
  for (int m = 0; m < N; ++m) o[m] = (_Float16)(r[m] * inv);
}

// ---------------- elementwise f16 subtract ----------------
__global__ void k_sub(const _Float16* __restrict__ a, const _Float16* __restrict__ b,
                      _Float16* __restrict__ o, int n)
{
  int i = blockIdx.x * blockDim.x + threadIdx.x;
  if (i < n) o[i] = (_Float16)((float)a[i] - (float)b[i]);
}

// ---------------- copy columns into concat buffer ----------------
__global__ void k_copy_cols(const _Float16* __restrict__ src, _Float16* __restrict__ dst,
                            int Cs, int Cd, int off, int M)
{
  int gid = blockIdx.x * blockDim.x + threadIdx.x;
  if (gid >= M * Cs) return;
  int c = gid % Cs, m = gid / Cs;
  dst[(size_t)m * Cd + off + c] = src[gid];
}

// ---------------- global max over points ----------------
__global__ void k_gmax(const _Float16* __restrict__ F, float* __restrict__ g,
                       int N, int C, int Bt)
{
  int gid = blockIdx.x * blockDim.x + threadIdx.x;
  if (gid >= Bt * C) return;
  int c = gid % C, b = gid / C;
  const _Float16* p = F + (size_t)b * N * C + c;
  float m = (float)p[0];
  for (int n = 1; n < N; ++n) { float v = (float)p[(size_t)n * C]; m = v > m ? v : m; }
  g[gid] = m;
}

// ---------------- FC head: 1024->512->256->40 (f32), one block per batch ----------------
__global__ void k_head(const float* __restrict__ g,
                       const float* w1, const float* bs1, const float* g1, const float* b1,
                       const float* w2, const float* bs2, const float* g2, const float* b2,
                       const float* w3, const float* bs3, float* __restrict__ out)
{
  int b = blockIdx.x, t = threadIdx.x;
  __shared__ float sg[1024], h1[512], h2[256];
  for (int i = t; i < 1024; i += 256) sg[i] = g[b * 1024 + i];
  __syncthreads();
  for (int o = t; o < 512; o += 256) {
    float s = bs1[o];
    for (int c = 0; c < 1024; ++c) s += sg[c] * w1[(size_t)o * 1024 + c];
    s = s * (g1[o] * RSQ) + b1[o];
    h1[o] = s > 0.f ? s : 0.f;
  }
  __syncthreads();
  for (int o = t; o < 256; o += 256) {
    float s = bs2[o];
    for (int c = 0; c < 512; ++c) s += h1[c] * w2[(size_t)o * 512 + c];
    s = s * (g2[o] * RSQ) + b2[o];
    h2[o] = s > 0.f ? s : 0.f;
  }
  __syncthreads();
  if (t < NCLS) {
    float s = bs3[t];
    for (int c = 0; c < 256; ++c) s += h2[c] * w3[(size_t)t * 256 + c];
    out[b * NCLS + t] = s;
  }
}

// ============================= host orchestration =============================
extern "C" void kernel_launch(void* const* d_in, const int* in_sizes, int n_in,
                              void* d_out, int out_size, void* d_ws, size_t ws_size,
                              hipStream_t stream)
{
  (void)in_sizes; (void)n_in; (void)out_size; (void)ws_size;
  auto F = [&](int i) -> const float* { return (const float*)d_in[i]; };

  size_t off = 0; char* base = (char*)d_ws;
  auto alloc = [&](size_t bytes) -> void* {
    void* p = base + off;
    off = (off + bytes + 255) & ~(size_t)255;
    return p;
  };

  // two big reusable regions (edge tensors / GEMM ping-pong / energy / attn)
  const size_t BIGSZ = (size_t)BB * S2 * KNB * 256 * sizeof(_Float16); // 134 MB
  _Float16* big0 = (_Float16*)alloc(BIGSZ);
  _Float16* big1 = (_Float16*)alloc(BIGSZ);

  _Float16* xyz_h = (_Float16*)alloc((size_t)BB*NPTS*3*2);
  _Float16* h1buf = (_Float16*)alloc((size_t)BB*NPTS*64*2);
  _Float16* feat  = (_Float16*)alloc((size_t)BB*NPTS*64*2);
  int*   idx1 = (int*)  alloc((size_t)BB*S1*4);
  int*   idx2 = (int*)  alloc((size_t)BB*S2*4);
  float* dist = (float*)alloc((size_t)BB*NPTS*4);
  int*   knn1 = (int*)  alloc((size_t)BB*S1*KNB*4);
  int*   knn2 = (int*)  alloc((size_t)BB*S2*KNB*4);
  float* xyz1 = (float*)alloc((size_t)BB*S1*3*4);
  _Float16* f1 = (_Float16*)alloc((size_t)BB*S1*128*2);
  _Float16* f2 = (_Float16*)alloc((size_t)BB*S2*256*2);
  _Float16* qb  = (_Float16*)alloc((size_t)BB*1024*64*2);
  _Float16* kb  = (_Float16*)alloc((size_t)BB*1024*64*2);
  _Float16* vb  = (_Float16*)alloc((size_t)BB*1024*256*2);
  _Float16* vT  = (_Float16*)alloc((size_t)BB*1024*256*2);
  _Float16* fsa = (_Float16*)alloc((size_t)BB*1024*256*2);
  _Float16* ub  = (_Float16*)alloc((size_t)BB*1024*256*2);
  _Float16* oo[4];
  for (int i = 0; i < 4; ++i) oo[i] = (_Float16*)alloc((size_t)BB*1024*256*2);
  _Float16* fin  = (_Float16*)alloc((size_t)BB*1024*1024*2);
  _Float16* fout = (_Float16*)alloc((size_t)BB*1024*1024*2);
  float* gbuf = (float*)alloc((size_t)BB*1024*4);

  auto cvtW = [&](int idx, int n) -> _Float16* {
    _Float16* p = (_Float16*)alloc((size_t)n * sizeof(_Float16));
    k_cvt<<<CDIV(n,256),256,0,stream>>>((const float*)d_in[idx], p, n);
    return p;
  };

  _Float16* w_s1l1 = cvtW(I_SG1_L1_W, 128*128);
  _Float16* w_s1l2 = cvtW(I_SG1_L2_W, 128*128);
  _Float16* w_s2l1 = cvtW(I_SG2_L1_W, 256*256);
  _Float16* w_s2l2 = cvtW(I_SG2_L2_W, 256*256);
  _Float16 *w_q[4], *w_k[4], *w_v[4], *w_pj[4];
  for (int i = 0; i < 4; ++i) {
    int IO = I_OA1 + 6*i;
    w_pj[i] = cvtW(IO+2, 256*256);
    w_k[i]  = cvtW(IO+3, 64*256);
    w_q[i]  = cvtW(IO+4, 64*256);
    w_v[i]  = cvtW(IO+5, 256*256);
  }
  _Float16* w_fuse = cvtW(I_FUSE_W, 1024*1024);

  // wave tile is 32x32, block covers 256 rows x 32 cols
  auto gemm = [&](const _Float16* X, const _Float16* W, _Float16* Y,
                  const float* g, const float* b, const _Float16* res,
                  int M, int Kd, int O, int relu,
                  int nb, long long sX, long long sW, long long sY, long long sRes) {
    dim3 gr((unsigned)CDIV(M,256), (unsigned)(O/32), (unsigned)nb);
    k_wmma_gemm<<<gr, 256, 0, stream>>>(X, W, Y, g, b, res, M, Kd, O, sX, sW, sY, sRes, relu);
  };

  // ---- base MLP: xyz -> feat [B,N,64] ----
  k_cvt<<<CDIV(BB*NPTS*3,256),256,0,stream>>>(F(I_XYZ), xyz_h, BB*NPTS*3);
  k_lin_small<<<CDIV(BB*NPTS*64,256),256,0,stream>>>(xyz_h, F(I_B1_W), F(I_B1_G), F(I_B1_B),
                                                     h1buf, BB*NPTS, 3, 64);
  k_lin_small<<<CDIV(BB*NPTS*64,256),256,0,stream>>>(h1buf, F(I_B2_W), F(I_B2_G), F(I_B2_B),
                                                     feat, BB*NPTS, 64, 64);

  // ---- SG1: 4096 -> 2048, 64 -> 128 ----
  k_fps<<<BB,256,0,stream>>>(F(I_XYZ), NPTS, S1, idx1, dist);
  k_knn<<<CDIV(BB*S1,128),128,0,stream>>>(F(I_XYZ), idx1, knn1, NPTS, S1, BB);
  k_edge<<<CDIV(BB*S1*KNB,256),256,0,stream>>>(feat, idx1, knn1, big0, NPTS, S1, 64, BB);
  gemm(big0, w_s1l1, big1, F(I_SG1_L1_G), F(I_SG1_L1_B), nullptr,
       BB*S1*KNB, 128, 128, 1, 1, 0,0,0,0);
  gemm(big1, w_s1l2, big0, F(I_SG1_L2_G), F(I_SG1_L2_B), nullptr,
       BB*S1*KNB, 128, 128, 1, 1, 0,0,0,0);
  k_maxk<<<CDIV(BB*S1*128,256),256,0,stream>>>(big0, f1, BB*S1, 128);
  k_gather3<<<CDIV(BB*S1,256),256,0,stream>>>(F(I_XYZ), idx1, xyz1, NPTS, S1, BB);

  // ---- SG2: 2048 -> 1024, 128 -> 256 ----
  k_fps<<<BB,256,0,stream>>>(xyz1, S1, S2, idx2, dist);
  k_knn<<<CDIV(BB*S2,128),128,0,stream>>>(xyz1, idx2, knn2, S1, S2, BB);
  k_edge<<<CDIV(BB*S2*KNB,256),256,0,stream>>>(f1, idx2, knn2, big0, S1, S2, 128, BB);
  gemm(big0, w_s2l1, big1, F(I_SG2_L1_G), F(I_SG2_L1_B), nullptr,
       BB*S2*KNB, 256, 256, 1, 1, 0,0,0,0);
  gemm(big1, w_s2l2, big0, F(I_SG2_L2_G), F(I_SG2_L2_B), nullptr,
       BB*S2*KNB, 256, 256, 1, 1, 0,0,0,0);
  k_maxk<<<CDIV(BB*S2*256,256),256,0,stream>>>(big0, f2, BB*S2, 256);

  // ---- 4x offset attention on [B,1024,256] ----
  const int M8 = BB * 1024;
  const _Float16* xin = f2;
  for (int i = 0; i < 4; ++i) {
    int IO = I_OA1 + 6*i;
    gemm(xin, w_q[i], qb, nullptr, nullptr, nullptr, M8, 256, 64, 0, 1, 0,0,0,0);
    gemm(xin, w_k[i], kb, nullptr, nullptr, nullptr, M8, 256, 64, 0, 1, 0,0,0,0);
    gemm(xin, w_v[i], vb, nullptr, nullptr, nullptr, M8, 256, 256, 0, 1, 0,0,0,0);
    k_transpose<<<CDIV(BB*1024*256,256),256,0,stream>>>(vb, vT, 1024, 256, BB);
    // energy[b] = q[b] @ k[b]^T : batched, big0 holds f16 energy [B,1024,1024]
    gemm(qb, kb, big0, nullptr, nullptr, nullptr, 1024, 64, 1024, 0,
         BB, 1024LL*64, 1024LL*64, 1024LL*1024, 0);
    k_col_softmax<<<CDIV(BB*1024,128),128,0,stream>>>(big0, (float*)big1, 1024, BB);
    k_row_renorm<<<CDIV(BB*1024,128),128,0,stream>>>((const float*)big1, big0, 1024, BB);
    // fsa[b] = attn[b] @ v[b] using vT so B fragments stay contiguous
    gemm(big0, vT, fsa, nullptr, nullptr, nullptr, 1024, 1024, 256, 0,
         BB, 1024LL*1024, 256LL*1024, 1024LL*256, 0);
    k_sub<<<CDIV(BB*1024*256,256),256,0,stream>>>(xin, fsa, ub, BB*1024*256);
    // o = relu(bn(u @ proj^T)) + x   (BN+ReLU+residual fused in epilogue)
    gemm(ub, w_pj[i], oo[i], F(IO+1), F(IO+0), xin, M8, 256, 256, 1, 1, 0,0,0,0);
    xin = oo[i];
  }

  // ---- concat + fuse LBR -> [B,1024,1024] ----
  for (int i = 0; i < 4; ++i)
    k_copy_cols<<<CDIV(M8*256,256),256,0,stream>>>(oo[i], fin, 256, 1024, 256*i, M8);
  gemm(fin, w_fuse, fout, F(I_FUSE_G), F(I_FUSE_B), nullptr, M8, 1024, 1024, 1, 1, 0,0,0,0);

  // ---- global max pool + FC head ----
  k_gmax<<<CDIV(BB*1024,256),256,0,stream>>>(fout, gbuf, 1024, 1024, BB);
  k_head<<<BB,256,0,stream>>>(gbuf,
                              F(I_FC1_W), F(I_FC1_BIAS), F(I_FC1_G), F(I_FC1_B),
                              F(I_FC2_W), F(I_FC2_BIAS), F(I_FC2_G), F(I_FC2_B),
                              F(I_FC3_W), F(I_FC3_BIAS), (float*)d_out);
}
